// ConditionalPreactivation_13048110645402
// MI455X (gfx1250) — matrix operand order repaired and checked by reading, not compiled, not run on hardware
//
#include <hip/hip_runtime.h>
#include <hip/hip_bf16.h>
#include <stdint.h>

// ---------------------------------------------------------------------------
// ConditionalPreactivation for MI455X (gfx1250), wave32 + WMMA bf16 +
// async global->LDS tile staging (double buffered).
//
// out[b,o] = sum_k bf[b,k] * sum_i a[b,i] * W[k,i,o]  + (bf @ bvec)[b,o]
// = one GEMM A'[B, K*DIN] x Wflat[K*DIN, DOUT], with the per-row bf scale
// folded into the accumulator between kouter segments (in registers).
// ---------------------------------------------------------------------------

typedef __attribute__((ext_vector_type(16))) __bf16       v16bf;
typedef __attribute__((ext_vector_type(8)))  float        v8f;
typedef __attribute__((ext_vector_type(8)))  unsigned int v8u;

#define B_    4096
#define DIN_  512
#define DOUT_ 512
#define DC_   64
#define K_    16
#define DB_   256
#define KDIN_ (K_ * DIN_)   // 8192 fused contraction length
#define NEG_  0.01f
#define EPS_  1e-5f

#define BM 128
#define BN 128
#define LDP 40              // LDS row stride in bf16: 80B, 16B-aligned, conflict-free

__device__ __forceinline__ float lrelu(float v) { return v >= 0.f ? v : NEG_ * v; }

__device__ __forceinline__ unsigned short f2bf(float f) {
    unsigned int u = __float_as_uint(f);
    u += 0x7FFFu + ((u >> 16) & 1u);   // round-to-nearest-even
    return (unsigned short)(u >> 16);
}

// gfx1250 async global->LDS copy, 16B per lane, tracked on ASYNCcnt.
__device__ __forceinline__ void async_b128(unsigned ldsAddr, const void* gptr) {
    asm volatile("global_load_async_to_lds_b128 %0, %1, off"
                 :: "v"(ldsAddr), "v"(gptr) : "memory");
}
__device__ __forceinline__ void wait_async0() {
    asm volatile("s_wait_asynccnt 0x0" ::: "memory");
}
__device__ __forceinline__ unsigned lds_off(const void* p) {
    // generic LDS address: low 32 bits are the LDS byte offset (ISA 10.2)
    return (unsigned)(unsigned long long)p;
}

// ---------------------------------------------------------------------------
// Kernel 1: LayerNorm + LeakyReLU, f32 -> bf16.  One wave32 per row.
// ---------------------------------------------------------------------------
__global__ __launch_bounds__(256) void ln_act_kernel(
    const float* __restrict__ x, const float* __restrict__ g,
    const float* __restrict__ bta, unsigned short* __restrict__ aB) {
    const int wid  = threadIdx.x >> 5;
    const int lane = threadIdx.x & 31;
    const int row  = blockIdx.x * 8 + wid;
    const float* xr = x + (size_t)row * DIN_;

    float v[16];
    float s = 0.f;
#pragma unroll
    for (int j = 0; j < 16; ++j) { v[j] = xr[lane + 32 * j]; s += v[j]; }
#pragma unroll
    for (int off = 16; off > 0; off >>= 1) s += __shfl_xor(s, off, 32);
    const float mu = s * (1.0f / DIN_);

    float vs = 0.f;
#pragma unroll
    for (int j = 0; j < 16; ++j) { float d = v[j] - mu; vs += d * d; }
#pragma unroll
    for (int off = 16; off > 0; off >>= 1) vs += __shfl_xor(vs, off, 32);
    const float rs = rsqrtf(vs * (1.0f / DIN_) + EPS_);

#pragma unroll
    for (int j = 0; j < 16; ++j) {
        const int i = lane + 32 * j;
        float xn = (v[j] - mu) * rs * g[i] + bta[i];
        aB[(size_t)row * DIN_ + i] = f2bf(lrelu(xn));
    }
}

// ---------------------------------------------------------------------------
// Kernel 2: W f32 [K*DIN, DOUT] -> bf16 TRANSPOSED  WbT[DOUT, K*DIN].
// Tiled 32x32 LDS transpose so both global read and write are coalesced.
// grid = (KDIN/32, DOUT/32), block = 256.
// ---------------------------------------------------------------------------
__global__ __launch_bounds__(256) void wtrans_kernel(
    const float* __restrict__ W, unsigned short* __restrict__ WbT) {
    __shared__ unsigned short tile[32][33];
    const int kk0 = blockIdx.x * 32;
    const int o0  = blockIdx.y * 32;
    const int tx  = threadIdx.x & 31;
    const int ty  = threadIdx.x >> 5;      // 8 row-groups
#pragma unroll
    for (int r = 0; r < 32; r += 8)
        tile[r + ty][tx] = f2bf(W[(size_t)(kk0 + r + ty) * DOUT_ + o0 + tx]);
    __syncthreads();
#pragma unroll
    for (int r = 0; r < 32; r += 8)
        WbT[(size_t)(o0 + r + ty) * KDIN_ + kk0 + tx] = tile[tx][r + ty];
}

// ---------------------------------------------------------------------------
// Kernel 3: basis_functions MLP  c[B,64] -> bf[B,16].  One block per row.
// ---------------------------------------------------------------------------
__global__ __launch_bounds__(256) void mlp_bf_kernel(
    const float* __restrict__ c,  const float* __restrict__ w1,
    const float* __restrict__ b1, const float* __restrict__ wr,
    const float* __restrict__ br, const float* __restrict__ w2,
    const float* __restrict__ b2, float* __restrict__ bf) {
    __shared__ float sC[DC_];
    __shared__ float sHL[DB_];
    __shared__ float sH3[DB_];
    const int b = blockIdx.x;
    const int j = threadIdx.x;

    if (j < DC_) sC[j] = c[(size_t)b * DC_ + j];
    __syncthreads();

    float h = b1[j];
    for (int i = 0; i < DC_; ++i) h += sC[i] * w1[i * DB_ + j];
    sHL[j] = lrelu(h);
    __syncthreads();

    float h2 = h + br[j];
    for (int i = 0; i < DB_; ++i) h2 += sHL[i] * wr[i * DB_ + j];
    sH3[j] = lrelu(h2);
    __syncthreads();

    const int wid = j >> 5, lane = j & 31;
#pragma unroll
    for (int kk = 0; kk < 2; ++kk) {
        const int k = wid + kk * 8;
        float p = 0.f;
#pragma unroll
        for (int t = 0; t < 8; ++t) {
            const int i = lane + 32 * t;
            p += sH3[i] * w2[i * K_ + k];
        }
#pragma unroll
        for (int off = 16; off > 0; off >>= 1) p += __shfl_xor(p, off, 32);
        if (lane == 0) bf[(size_t)b * K_ + k] = p + b2[k];
    }
}

// ---------------------------------------------------------------------------
// Kernel 4: fused WMMA GEMM, async double-buffered LDS staging.
// Block = 256 threads (8 waves); tile = 128(M) x 128(N); BK = 32.
// Wave w -> 32x64 sub-tile (2x4 v_wmma_f32_16x16x32_bf16 tiles):
//   8 WMMAs per wave per barrier; each A frag feeds 4 WMMAs.
// ---------------------------------------------------------------------------
__global__ __launch_bounds__(256) void fused_gemm_kernel(
    const unsigned short* __restrict__ aB,    // [B, DIN] bf16
    const unsigned short* __restrict__ WbT,   // [DOUT, K*DIN] bf16 (transposed)
    const float* __restrict__ bf,             // [B, K]
    const float* __restrict__ bvec,           // [K, DOUT]
    float* __restrict__ out) {                // [B, DOUT]
    __shared__ unsigned short sA[2][BM][LDP];   // 20 KB
    __shared__ unsigned short sBT[2][BN][LDP];  // 20 KB
    __shared__ float sBF[BM][K_];               //  8 KB
    __shared__ float sBV[K_][BN];               //  8 KB   (total 56 KB)

    const int tid     = threadIdx.x;
    const int rowBase = blockIdx.y * BM;
    const int colBase = blockIdx.x * BN;

    for (int t = tid; t < BM * K_; t += 256)
        sBF[t >> 4][t & 15] = bf[(size_t)(rowBase + (t >> 4)) * K_ + (t & 15)];
    for (int t = tid; t < K_ * BN; t += 256)
        sBV[t >> 7][t & 127] = bvec[(size_t)(t >> 7) * DOUT_ + colBase + (t & 127)];

    const int wid   = tid >> 5, lane = tid & 31;
    const int hf    = lane >> 4;        // lane half (K split per ISA layout)
    const int l16   = lane & 15;        // M for A frags, N for B/C frags
    const int mBase = (wid & 3) * 32;   // 4 waves tile M
    const int nBase = (wid >> 2) * 64;  // 2 waves tile N, 64 cols each

    // per-thread staging coordinates (16B chunks); 512 chunks per 128x32 tile
    const int r0 = tid >> 2,         c0 = (tid & 3) << 3;
    const int r1 = (tid + 256) >> 2, c1 = c0;

    // issue async copies for k-step `ks` into buffer `bsel`
    auto issue_tiles = [&](int bsel, int ks) {
        const int kout = ks >> 4;
        const int i0   = (ks & 15) << 5;
        const size_t kkOff = (size_t)kout * DIN_ + i0;
        async_b128(lds_off(&sA[bsel][r0][c0]),
                   &aB[(size_t)(rowBase + r0) * DIN_ + i0 + c0]);
        async_b128(lds_off(&sA[bsel][r1][c1]),
                   &aB[(size_t)(rowBase + r1) * DIN_ + i0 + c1]);
        async_b128(lds_off(&sBT[bsel][r0][c0]),
                   &WbT[(size_t)(colBase + r0) * KDIN_ + kkOff + c0]);
        async_b128(lds_off(&sBT[bsel][r1][c1]),
                   &WbT[(size_t)(colBase + r1) * KDIN_ + kkOff + c1]);
    };

    v8f facc[2][4] = {};
    v8f pacc[2][4] = {};

    int buf = 0;
    issue_tiles(0, 0);
    wait_async0();
    __syncthreads();   // also covers sBF/sBV staging

    for (int ks = 0; ks < 16 * K_; ++ks) {
        if (ks + 1 < 16 * K_) issue_tiles(buf ^ 1, ks + 1);

        // ---- fragments per ISA 7.12.2 bf16 layouts ----
        v16bf aF[2], bF[4];
#pragma unroll
        for (int mi = 0; mi < 2; ++mi) {
            v8u u;
            const int row = mBase + mi * 16 + l16;
#pragma unroll
            for (int v = 0; v < 8; ++v) {
                // A 16x32: K pair base = (v/4)*16 + half*8 + (v%4)*2
                const int k0 = ((v >> 2) << 4) + (hf << 3) + ((v & 3) << 1);
                u[v] = *(const unsigned int*)&sA[buf][row][k0];
            }
            aF[mi] = __builtin_bit_cast(v16bf, u);
        }
#pragma unroll
        for (int ni = 0; ni < 4; ++ni) {
            v8u u;
            const int col = nBase + ni * 16 + l16;
#pragma unroll
            for (int v = 0; v < 8; ++v) {
                // B 32x16: K pair base = (v/4)*16 + half*8 + (v%4)*2, N = lane&15
                const int k0 = ((v >> 2) << 4) + (hf << 3) + ((v & 3) << 1);
                u[v] = *(const unsigned int*)&sBT[buf][col][k0];
            }
            bF[ni] = __builtin_bit_cast(v16bf, u);
        }
#pragma unroll
        for (int mi = 0; mi < 2; ++mi)
#pragma unroll
            for (int ni = 0; ni < 4; ++ni)
                pacc[mi][ni] = __builtin_amdgcn_wmma_f32_16x16x32_bf16(
                    false, aF[mi], false, bF[ni], (short)0, pacc[mi][ni],
                    false, false);

        // ---- end of a kouter segment: fold with per-row bf scale ----
        if ((ks & 15) == 15) {
            const int kout = ks >> 4;
#pragma unroll
            for (int mi = 0; mi < 2; ++mi) {
#pragma unroll
                for (int v = 0; v < 8; ++v) {
                    const int row = mBase + mi * 16 + v + 8 * hf;  // C/D: M=v+8*half
                    const float s = sBF[row][kout];
#pragma unroll
                    for (int ni = 0; ni < 4; ++ni) {
                        facc[mi][ni][v] += s * pacc[mi][ni][v];
                        pacc[mi][ni][v] = 0.f;
                    }
                }
            }
        }

        wait_async0();      // our next-tile copies have overlapped the WMMAs
        __syncthreads();    // all waves' copies visible, all reads of buf done
        buf ^= 1;
    }

    // ---- epilogue: + bf @ bvec, store ----
#pragma unroll
    for (int mi = 0; mi < 2; ++mi)
#pragma unroll
        for (int ni = 0; ni < 4; ++ni) {
            const int ncol = nBase + ni * 16 + l16;
#pragma unroll
            for (int v = 0; v < 8; ++v) {
                const int row = mBase + mi * 16 + v + 8 * hf;
                float acc = facc[mi][ni][v];
#pragma unroll
                for (int k = 0; k < K_; ++k) acc += sBF[row][k] * sBV[k][ncol];
                out[(size_t)(rowBase + row) * DOUT_ + colBase + ncol] = acc;
            }
        }
}

// ---------------------------------------------------------------------------
extern "C" void kernel_launch(void* const* d_in, const int* in_sizes, int n_in,
                              void* d_out, int out_size, void* d_ws,
                              size_t ws_size, hipStream_t stream) {
    (void)in_sizes; (void)n_in; (void)out_size; (void)ws_size;

    const float* x    = (const float*)d_in[0];
    const float* c    = (const float*)d_in[1];
    const float* g    = (const float*)d_in[2];
    const float* bta  = (const float*)d_in[3];
    const float* w1   = (const float*)d_in[4];
    const float* b1   = (const float*)d_in[5];
    const float* wr   = (const float*)d_in[6];
    const float* br   = (const float*)d_in[7];
    const float* w2   = (const float*)d_in[8];
    const float* b2   = (const float*)d_in[9];
    const float* W    = (const float*)d_in[10];
    const float* bvec = (const float*)d_in[11];
    float* out = (float*)d_out;

    char* ws = (char*)d_ws;
    unsigned short* aB  = (unsigned short*)ws;                            // 4 MB
    unsigned short* WbT = (unsigned short*)(ws + (size_t)B_ * DIN_ * 2);  // 8 MB
    float* bfv = (float*)(ws + (size_t)B_ * DIN_ * 2 +
                          (size_t)KDIN_ * DOUT_ * 2);                     // 256 KB

    ln_act_kernel<<<B_ / 8, 256, 0, stream>>>(x, g, bta, aB);
    dim3 tg(KDIN_ / 32, DOUT_ / 32);
    wtrans_kernel<<<tg, 256, 0, stream>>>(W, WbT);
    mlp_bf_kernel<<<B_, 256, 0, stream>>>(c, w1, b1, wr, br, w2, b2, bfv);

    dim3 grid(DOUT_ / BN, B_ / BM);
    fused_gemm_kernel<<<grid, 256, 0, stream>>>(aB, WbT, bfv, bvec, out);
}
